// GAT_10084583211425
// MI455X (gfx1250) — compile-verified, matrix operand order checked by reading, not run on hardware
//
#include <hip/hip_runtime.h>
#include <hip/hip_bf16.h>
#include <cmath>

// ---------------------------------------------------------------------------
// Problem constants (from the reference)
// ---------------------------------------------------------------------------
#define NN   50000LL
#define EE   1000000LL
#define EF   1050000LL   // E + N self loops
#define INF_ 384
#define F12  256         // HEADS*H1 == HEADS*H2
#define F3   384         // HEADS3*H3
#define HH   4
#define HH3  6
#define CH   64
#define GG   64

#define GAT_ASYNC_LDS 1   // use global_load_async_to_lds_b128 for weight staging
#define LDS_PAD 8         // pad B rows in LDS: (K+8)*2B row stride -> conflict-free b128 reads

typedef __attribute__((ext_vector_type(16))) _Float16 v16h;
typedef __attribute__((ext_vector_type(8)))  _Float16 v8h;
typedef __attribute__((ext_vector_type(8)))  float    v8f;

// ---------------------------------------------------------------------------
// WMMA GEMM with LDS-staged weights:
//   C[M,Nout] = A[M,K] * Bt[Nout,K]^T   (f16 in, f32 out)
// Block = 256 threads (8 waves). Block owns a 64-column slice of Bt, staged
// into LDS once (async global->LDS). Each wave computes a 16x64 output strip:
// one A fragment is reused by 4 v_wmma_f32_16x16x32_f16 per K-step.
// ---------------------------------------------------------------------------
__global__ void gat_gemm_wmma_lds(const _Float16* __restrict__ A,
                                  const _Float16* __restrict__ Bt,
                                  float* __restrict__ C,
                                  int M, int K, int Nout) {
  __shared__ __align__(16) _Float16 sB[64 * (384 + LDS_PAD)];

  const int tid  = threadIdx.x;
  const int ldk  = K + LDS_PAD;
  const int col0 = blockIdx.y * 64;

  // ---- stage Bt[col0..col0+63][0..K) into LDS (16B chunks, all lanes active)
  const int kq = K >> 3;                 // 16B chunks per row
  for (int idx = tid; idx < 64 * kq; idx += 256) {
    int row = idx / kq;
    int kk  = (idx - row * kq) << 3;     // in halves
    const uint4* gp = (const uint4*)(Bt + (size_t)(col0 + row) * K + kk);
#if GAT_ASYNC_LDS
    unsigned ldsoff = (unsigned)(uintptr_t)&sB[row * ldk + kk];  // low 32b = LDS offset
    asm volatile("global_load_async_to_lds_b128 %0, %1, off"
                 :: "v"(ldsoff), "v"(gp) : "memory");
#else
    *(uint4*)&sB[row * ldk + kk] = *gp;
#endif
  }
#if GAT_ASYNC_LDS
  asm volatile("s_wait_asynccnt 0x0" ::: "memory");
#endif
  __syncthreads();

  // ---- per-wave 16x64 strip
  const int wv    = tid >> 5;
  const int lane  = tid & 31;
  const int r     = lane & 15;
  const int kb    = (lane >> 4) << 3;    // 0 / 8 (halves)
  const int mtile = blockIdx.x * 8 + wv;
  if (mtile * 16 >= M) return;           // no barriers after this point

  const _Float16* ap  = A + (size_t)(mtile * 16 + r) * K + kb;
  const _Float16* bp0 = &sB[(0 * 16 + r) * ldk + kb];
  const _Float16* bp1 = &sB[(1 * 16 + r) * ldk + kb];
  const _Float16* bp2 = &sB[(2 * 16 + r) * ldk + kb];
  const _Float16* bp3 = &sB[(3 * 16 + r) * ldk + kb];

  v8f acc0 = {}, acc1 = {}, acc2 = {}, acc3 = {};
#pragma unroll 2
  for (int k = 0; k < K; k += 32) {
    v8h a0 = *(const v8h*)(ap + k);
    v8h a1 = *(const v8h*)(ap + k + 16);
    v16h a = __builtin_shufflevector(a0, a1, 0,1,2,3,4,5,6,7,8,9,10,11,12,13,14,15);

    v8h c0 = *(const v8h*)(bp0 + k);
    v8h c1 = *(const v8h*)(bp0 + k + 16);
    v16h b = __builtin_shufflevector(c0, c1, 0,1,2,3,4,5,6,7,8,9,10,11,12,13,14,15);
    acc0 = __builtin_amdgcn_wmma_f32_16x16x32_f16(false, a, false, b, (short)0, acc0, false, false);

    c0 = *(const v8h*)(bp1 + k);
    c1 = *(const v8h*)(bp1 + k + 16);
    b  = __builtin_shufflevector(c0, c1, 0,1,2,3,4,5,6,7,8,9,10,11,12,13,14,15);
    acc1 = __builtin_amdgcn_wmma_f32_16x16x32_f16(false, a, false, b, (short)0, acc1, false, false);

    c0 = *(const v8h*)(bp2 + k);
    c1 = *(const v8h*)(bp2 + k + 16);
    b  = __builtin_shufflevector(c0, c1, 0,1,2,3,4,5,6,7,8,9,10,11,12,13,14,15);
    acc2 = __builtin_amdgcn_wmma_f32_16x16x32_f16(false, a, false, b, (short)0, acc2, false, false);

    c0 = *(const v8h*)(bp3 + k);
    c1 = *(const v8h*)(bp3 + k + 16);
    b  = __builtin_shufflevector(c0, c1, 0,1,2,3,4,5,6,7,8,9,10,11,12,13,14,15);
    acc3 = __builtin_amdgcn_wmma_f32_16x16x32_f16(false, a, false, b, (short)0, acc3, false, false);
  }

  const int row0 = mtile * 16 + ((lane >> 4) << 3);
  float* cp = C + (size_t)row0 * Nout + col0 + r;
#pragma unroll
  for (int i = 0; i < 8; i++) {
    float* cr = cp + (size_t)i * Nout;
    cr[0]  = acc0[i];
    cr[16] = acc1[i];
    cr[32] = acc2[i];
    cr[48] = acc3[i];
  }
}

// ---------------------------------------------------------------------------
// Helpers
// ---------------------------------------------------------------------------
__global__ void gat_fill(float* __restrict__ p, float v, long long n) {
  long long t = (long long)blockIdx.x * blockDim.x + threadIdx.x;
  if (t < n) p[t] = v;
}

__global__ void gat_f32_to_f16(const float* __restrict__ in,
                               _Float16* __restrict__ out, long long n) {
  long long t = (long long)blockIdx.x * blockDim.x + threadIdx.x;
  if (t < n) out[t] = (_Float16)in[t];
}

// Wt[n*K + k] = (f16) W[k*Nout + n]
__global__ void gat_w_transpose(const float* __restrict__ W,
                                _Float16* __restrict__ Wt, int K, int Nout) {
  int t = blockIdx.x * blockDim.x + threadIdx.x;
  if (t >= K * Nout) return;
  int k = t / Nout, n = t - k * Nout;
  Wt[(size_t)n * K + k] = (_Float16)W[t];
}

// Build src/dst (with self loops appended) and write ei_full (as float) to the
// three output slots.
__global__ void gat_build_edges(const int* __restrict__ ei,
                                int* __restrict__ src, int* __restrict__ dst,
                                float* o1, float* o2, float* o3) {
  long long t = (long long)blockIdx.x * blockDim.x + threadIdx.x;
  if (t >= EF) return;
  int s = (t < EE) ? ei[t]      : (int)(t - EE);
  int d = (t < EE) ? ei[EE + t] : (int)(t - EE);
  src[t] = s; dst[t] = d;
  float fs = (float)s, fd = (float)d;
  o1[t] = fs; o1[EF + t] = fd;
  o2[t] = fs; o2[EF + t] = fd;
  o3[t] = fs; o3[EF + t] = fd;
}

// a_s / a_d: per (node, head) dot of h with attention vectors
__global__ void gat_att_coeff(const float* __restrict__ h,
                              const float* __restrict__ atts,
                              const float* __restrict__ attd,
                              float* __restrict__ as_, float* __restrict__ ad_,
                              long long nn, int heads) {
  long long t = (long long)blockIdx.x * blockDim.x + threadIdx.x;
  if (t >= nn * heads) return;
  long long node = t / heads;
  int hd = (int)(t - node * heads);
  const float* hp = h + (size_t)node * heads * CH + (size_t)hd * CH;
  float ss = 0.f, sd = 0.f;
  for (int c = 0; c < CH; c++) {
    float v = hp[c];
    ss += v * atts[hd * CH + c];
    sd += v * attd[hd * CH + c];
  }
  as_[t] = ss; ad_[t] = sd;
}

__device__ __forceinline__ void atomicMaxFloat(float* addr, float val) {
  if (val >= 0.f) atomicMax((int*)addr, __float_as_int(val));
  else            atomicMin((unsigned int*)addr, __float_as_uint(val));
}

// pass 1: logits + per-destination running max
__global__ void gat_edge_logit(const int* __restrict__ src, const int* __restrict__ dst,
                               const float* __restrict__ as_, const float* __restrict__ ad_,
                               float* __restrict__ eb, float* __restrict__ mm, int heads) {
  long long t = (long long)blockIdx.x * blockDim.x + threadIdx.x;
  if (t >= EF * heads) return;
  long long e = t / heads;
  int hd = (int)(t - e * heads);
  int s = src[e], d = dst[e];
  float v = as_[(long long)s * heads + hd] + ad_[(long long)d * heads + hd];
  v = (v > 0.f) ? v : 0.2f * v;                 // leaky_relu(0.2)
  eb[t] = v;
  atomicMaxFloat(&mm[(long long)d * heads + hd], v);
}

// pass 2: exp(e - m[dst]) and per-destination sum
__global__ void gat_edge_expsum(const int* __restrict__ dst,
                                float* __restrict__ eb, const float* __restrict__ mm,
                                float* __restrict__ den, int heads) {
  long long t = (long long)blockIdx.x * blockDim.x + threadIdx.x;
  if (t >= EF * heads) return;
  long long e = t / heads;
  int hd = (int)(t - e * heads);
  float m = mm[(long long)dst[e] * heads + hd];
  m = (m > -3.0e38f) ? m : 0.0f;                // finite guard (matches ref)
  float ex = expf(eb[t] - m);
  eb[t] = ex;
  atomicAdd(&den[(long long)dst[e] * heads + hd], ex);
}

// pass 3: alpha = ex/(den+eps) -> output + reuse buffer
__global__ void gat_edge_alpha(const int* __restrict__ dst,
                               float* __restrict__ eb, const float* __restrict__ den,
                               float* __restrict__ alpha_out, int heads) {
  long long t = (long long)blockIdx.x * blockDim.x + threadIdx.x;
  if (t >= EF * heads) return;
  long long e = t / heads;
  int hd = (int)(t - e * heads);
  float a = eb[t] / (den[(long long)dst[e] * heads + hd] + 1e-16f);
  alpha_out[t] = a;
  eb[t] = a;
}

// pass 4: out[dst] += h[src] * alpha   (4 channels / thread, float4 gather)
__global__ void gat_edge_msg(const int* __restrict__ src, const int* __restrict__ dst,
                             const float* __restrict__ alpha, const float* __restrict__ h,
                             float* __restrict__ out, int heads, int F) {
  long long t = (long long)blockIdx.x * blockDim.x + threadIdx.x;
  int fq = F >> 2;
  if (t >= EF * fq) return;
  long long e = t / fq;
  int c = (int)(t - e * fq) << 2;
  int hd = c / CH;                                 // c..c+3 share a head (CH=64)
  float al = alpha[e * heads + hd];
  const float4 hv = *(const float4*)(h + (size_t)src[e] * F + c);
  float* op = out + (size_t)dst[e] * F + c;
  atomicAdd(op + 0, hv.x * al);
  atomicAdd(op + 1, hv.y * al);
  atomicAdd(op + 2, hv.z * al);
  atomicAdd(op + 3, hv.w * al);
}

// bias + LayerNorm + ReLU, emit f16 for the next WMMA GEMM. One wave per node.
__global__ void gat_bias_ln_relu(const float* __restrict__ acc, const float* __restrict__ bias,
                                 const float* __restrict__ g, const float* __restrict__ bet,
                                 _Float16* __restrict__ xh, long long nn, int F) {
  long long node = (long long)blockIdx.x * (blockDim.x >> 5) + (threadIdx.x >> 5);
  if (node >= nn) return;
  int lane = threadIdx.x & 31;
  int per  = F >> 5;                               // 8 for F=256
  const float* ap = acc + (size_t)node * F + lane * per;
  float v[8];
  float s = 0.f;
  for (int i = 0; i < per; i++) { v[i] = ap[i] + bias[lane * per + i]; s += v[i]; }
  for (int o = 16; o; o >>= 1) s += __shfl_xor(s, o, 32);
  float mu = s / F;
  float q = 0.f;
  for (int i = 0; i < per; i++) { float d = v[i] - mu; q += d * d; }
  for (int o = 16; o; o >>= 1) q += __shfl_xor(q, o, 32);
  float inv = rsqrtf(q / F + 1e-5f);
  _Float16* xp = xh + (size_t)node * F + lane * per;
  for (int i = 0; i < per; i++) {
    float xn = (v[i] - mu) * inv * g[lane * per + i] + bet[lane * per + i];
    xn = (xn > 0.f) ? xn : 0.f;
    xp[i] = (_Float16)xn;
  }
}

// layer3: relu(acc+b3) + residual x, scatter into per-graph pool sums/counts
__global__ void gat_layer3_post(const float* __restrict__ acc, const float* __restrict__ b3,
                                const float* __restrict__ x, const int* __restrict__ batch,
                                float* __restrict__ pool, float* __restrict__ cnt) {
  long long t = (long long)blockIdx.x * blockDim.x + threadIdx.x;
  if (t >= NN * F3) return;
  long long node = t / F3;
  int f = (int)(t - node * F3);
  float x3 = acc[t] + b3[f];
  x3 = (x3 > 0.f) ? x3 : 0.f;
  float xr = x[t] + x3;
  int b = batch[node];
  atomicAdd(&pool[(size_t)b * F3 + f], xr);
  if (f == 0) atomicAdd(&cnt[b], 1.0f);
}

// mean-pool + fc1(relu) + fc2 -> out[64,8]
__global__ void gat_mlp(const float* __restrict__ pool, const float* __restrict__ cnt,
                        const float* __restrict__ w1, const float* __restrict__ b1,
                        const float* __restrict__ w2, const float* __restrict__ b2,
                        float* __restrict__ outp) {
  int g = threadIdx.x;
  if (g >= GG) return;
  float c = cnt[g]; c = (c > 1.f) ? c : 1.f;
  float inv = 1.0f / c;
  const float* pg = pool + (size_t)g * F3;
  float hid[32];
  for (int j = 0; j < 32; j++) {
    float a = b1[j];
    for (int k = 0; k < F3; k++) a += (pg[k] * inv) * w1[k * 32 + j];
    hid[j] = (a > 0.f) ? a : 0.f;
  }
  for (int o = 0; o < 8; o++) {
    float a = b2[o];
    for (int j = 0; j < 32; j++) a += hid[j] * w2[j * 8 + o];
    outp[g * 8 + o] = a;
  }
}

// ---------------------------------------------------------------------------
// Host side
// ---------------------------------------------------------------------------
static inline unsigned cdivll(long long a, long long b) { return (unsigned)((a + b - 1) / b); }

extern "C" void kernel_launch(void* const* d_in, const int* in_sizes, int n_in,
                              void* d_out, int out_size, void* d_ws, size_t ws_size,
                              hipStream_t stream) {
  const float* x     = (const float*)d_in[0];
  const int*   ei    = (const int*)  d_in[1];
  const int*   batch = (const int*)  d_in[2];
  const float* W1    = (const float*)d_in[3];
  const float* atts1 = (const float*)d_in[4];
  const float* attd1 = (const float*)d_in[5];
  const float* b1    = (const float*)d_in[6];
  const float* ln1g  = (const float*)d_in[7];
  const float* ln1b  = (const float*)d_in[8];
  const float* W2    = (const float*)d_in[9];
  const float* atts2 = (const float*)d_in[10];
  const float* attd2 = (const float*)d_in[11];
  const float* b2    = (const float*)d_in[12];
  const float* ln2g  = (const float*)d_in[13];
  const float* ln2b  = (const float*)d_in[14];
  const float* W3    = (const float*)d_in[15];
  const float* atts3 = (const float*)d_in[16];
  const float* attd3 = (const float*)d_in[17];
  const float* b3    = (const float*)d_in[18];
  const float* fc1w  = (const float*)d_in[19];
  const float* fc1b  = (const float*)d_in[20];
  const float* fc2w  = (const float*)d_in[21];
  const float* fc2b  = (const float*)d_in[22];

  float* outf = (float*)d_out;
  // d_out layout (floats): out[512] | ei1[2*EF] | a1[EF*4] | ei2 | a2[EF*4] | ei3 | a3[EF*6]
  const long long O_EI1 = 512;
  const long long O_A1  = O_EI1 + 2 * EF;
  const long long O_EI2 = O_A1 + EF * HH;
  const long long O_A2  = O_EI2 + 2 * EF;
  const long long O_EI3 = O_A2 + EF * HH;
  const long long O_A3  = O_EI3 + 2 * EF;

  // workspace carve-up
  char* ws = (char*)d_ws;
  size_t off = 0;
  auto take = [&](size_t bytes) -> char* {
    char* r = ws + off;
    off += (bytes + 255) & ~(size_t)255;
    return r;
  };
  _Float16* XH  = (_Float16*)take(NN * INF_ * 2);     // f16 of x
  _Float16* W1T = (_Float16*)take((size_t)INF_ * F12 * 2);
  _Float16* W2T = (_Float16*)take((size_t)F12 * F12 * 2);
  _Float16* W3T = (_Float16*)take((size_t)F12 * F3  * 2);
  float*    HB  = (float*)   take(NN * F3 * 4);       // gemm output h
  float*    OB  = (float*)   take(NN * F3 * 4);       // attention output accumulator
  _Float16* XH2 = (_Float16*)take(NN * F12 * 2);      // f16 activations (layers 2,3 input)
  float*    AS  = (float*)   take(NN * HH3 * 4);
  float*    AD  = (float*)   take(NN * HH3 * 4);
  float*    MM  = (float*)   take(NN * HH3 * 4);
  float*    DEN = (float*)   take(NN * HH3 * 4);
  float*    EB  = (float*)   take(EF * HH3 * 4);
  int*      SRC = (int*)     take(EF * 4);
  int*      DST = (int*)     take(EF * 4);
  float*    PL  = (float*)   take((size_t)GG * F3 * 4);
  float*    CT  = (float*)   take(GG * 4);
  (void)ws_size; (void)in_sizes; (void)n_in; (void)out_size;

  const int TB = 256;

  auto gemm = [&](const _Float16* A, const _Float16* Bt, float* C, int M, int K, int Nout) {
    dim3 grid(cdivll(M / 16, 8), Nout / 64, 1);
    gat_gemm_wmma_lds<<<grid, 256, 0, stream>>>(A, Bt, C, M, K, Nout);
  };

  auto attention = [&](int heads, int F, const float* atts, const float* attd,
                       float* alpha_out) {
    long long nh = NN * heads, eh = EF * heads;
    gat_att_coeff <<<cdivll(nh, TB), TB, 0, stream>>>(HB, atts, attd, AS, AD, NN, heads);
    gat_fill      <<<cdivll(nh, TB), TB, 0, stream>>>(MM, -INFINITY, nh);
    gat_fill      <<<cdivll(nh, TB), TB, 0, stream>>>(DEN, 0.0f, nh);
    gat_fill      <<<cdivll(NN * F, TB), TB, 0, stream>>>(OB, 0.0f, NN * F);
    gat_edge_logit <<<cdivll(eh, TB), TB, 0, stream>>>(SRC, DST, AS, AD, EB, MM, heads);
    gat_edge_expsum<<<cdivll(eh, TB), TB, 0, stream>>>(DST, EB, MM, DEN, heads);
    gat_edge_alpha <<<cdivll(eh, TB), TB, 0, stream>>>(DST, EB, DEN, alpha_out, heads);
    long long mt = EF * (F >> 2);
    gat_edge_msg  <<<cdivll(mt, TB), TB, 0, stream>>>(SRC, DST, EB, HB, OB, heads, F);
  };

  // ---- prep ----
  gat_f32_to_f16 <<<cdivll(NN * INF_, TB), TB, 0, stream>>>(x, XH, NN * INF_);
  gat_w_transpose<<<cdivll((long long)INF_ * F12, TB), TB, 0, stream>>>(W1, W1T, INF_, F12);
  gat_w_transpose<<<cdivll((long long)F12 * F12, TB), TB, 0, stream>>>(W2, W2T, F12, F12);
  gat_w_transpose<<<cdivll((long long)F12 * F3, TB), TB, 0, stream>>>(W3, W3T, F12, F3);
  gat_build_edges<<<cdivll(EF, TB), TB, 0, stream>>>(ei, SRC, DST,
                                                     outf + O_EI1, outf + O_EI2, outf + O_EI3);

  // ---- layer 1 ----
  gemm(XH, W1T, HB, (int)NN, INF_, F12);
  attention(HH, F12, atts1, attd1, outf + O_A1);
  gat_bias_ln_relu<<<cdivll(NN, 8), 256, 0, stream>>>(OB, b1, ln1g, ln1b, XH2, NN, F12);

  // ---- layer 2 ----
  gemm(XH2, W2T, HB, (int)NN, F12, F12);
  attention(HH, F12, atts2, attd2, outf + O_A2);
  gat_bias_ln_relu<<<cdivll(NN, 8), 256, 0, stream>>>(OB, b2, ln2g, ln2b, XH2, NN, F12);

  // ---- layer 3 ----
  gemm(XH2, W3T, HB, (int)NN, F12, F3);
  attention(HH3, F3, atts3, attd3, outf + O_A3);

  // ---- pooling + MLP head ----
  gat_fill<<<cdivll((long long)GG * F3, TB), TB, 0, stream>>>(PL, 0.0f, (long long)GG * F3);
  gat_fill<<<1, GG, 0, stream>>>(CT, 0.0f, GG);
  gat_layer3_post<<<cdivll(NN * F3, TB), TB, 0, stream>>>(OB, b3, x, batch, PL, CT);
  gat_mlp<<<1, GG, 0, stream>>>(PL, CT, fc1w, fc1b, fc2w, fc2b, outf);
}
